// DeltaNetBlock_54159537602962
// MI455X (gfx1250) — matrix operand order repaired and checked by reading, not compiled, not run on hardware
//
#include <hip/hip_runtime.h>
#include <hip/hip_bf16.h>
#include <math.h>
#include <stdint.h>

typedef __attribute__((ext_vector_type(16))) _Float16 v16h;
typedef __attribute__((ext_vector_type(8)))  _Float16 v8h;
typedef __attribute__((ext_vector_type(8)))  float    v8f;

#define Bn  8
#define Ln  1024
#define Hn  1024
#define En  2048
#define BLn (Bn*Ln)

__device__ __forceinline__ float sigmoid_f(float x){ return 1.0f/(1.0f+__expf(-x)); }
__device__ __forceinline__ float silu_f(float x){ return x*sigmoid_f(x); }
__device__ __forceinline__ float gelu_f(float x){ return 0.5f*x*(1.0f+erff(x*0.70710678118654752f)); }

// CDNA5 async memory->LDS copy (GLOBAL_LOAD_ASYNC_TO_LDS_B128, ASYNCcnt).
// GV addressing mode: 64-bit global address in a VGPR pair, LDS byte address
// (low 32 bits of the generic shared pointer) in a VGPR.
__device__ __forceinline__ void async_copy_b128(void* lds_dst, const void* gsrc)
{
    uint32_t l = (uint32_t)(uintptr_t)lds_dst;
    uint64_t g = (uint64_t)(uintptr_t)gsrc;
    asm volatile("global_load_async_to_lds_b128 %0, %1, off"
                 :: "v"(l), "v"(g) : "memory");
}
#define WAIT_ASYNC(N) asm volatile("s_wait_asynccnt %0" :: "n"(N) : "memory")

// ---------------------------------------------------------------------------
// Generic WMMA GEMM: C[m,n] = epilogue( sum_k A[m*lda+k] * B[n*ldb+k] )
// 128x128 block tile, 8 waves (2 in M x 4 in N), each wave = 4x2 accum tiles
// of v_wmma_f32_16x16x32_f16.  K stride is always 1 for every operand (data
// layouts are pre-arranged) -> all staging is aligned 16B transfers.
// Double-buffered LDS, filled with global_load_async_to_lds_b128 so the next
// tile's fill overlaps the current tile's 8 WMMAs (ASYNCcnt pipelining).
// Epilogues (template EPI):
//   0: raw f16 store                       (state)
//   1: +bias, silu(n<4096)/gelu  -> f16    (qkv)
//   2: +bias + rowv[m]*colv[n], sigmoid*0.9+0.1 -> f16   (beta)
//   3: * (*scale)               -> f32     (attention out)
//   4: +bias + resid            -> f32     (w_out + residual)
//   5: +bias, gelu              -> f16     (mlp1)
//   6: +bias + resid            -> f32     (mlp2 + residual -> d_out)
// ---------------------------------------------------------------------------
#define TM 128
#define TN 128
#define TK 32
#define LSTR 40   // f16 elems per LDS row: 32 + 8 pad => 80B, keeps 16B align

template<int EPI>
__global__ __launch_bounds__(256)
void gemm_wmma(const _Float16* __restrict__ A, long lda, long bsA,
               const _Float16* __restrict__ Bw, long ldb, long bsB,
               int K,
               const float* __restrict__ bias,
               const float* __restrict__ rowv,
               const float* __restrict__ colv,
               const float* __restrict__ scale,
               const float* __restrict__ resid, long ldr,
               void* __restrict__ Cout, long ldc, long bsC)
{
    __shared__ _Float16 As[2][TM*LSTR];
    __shared__ _Float16 Bs[2][TN*LSTR];

    const int tid  = threadIdx.x;
    const int lane = tid & 31;
    const int wave = tid >> 5;
    const int wm   = wave & 1;     // 2 waves along M (64 rows each)
    const int wn   = wave >> 1;    // 4 waves along N (32 cols each)
    const int z    = blockIdx.z;
    const long tm  = (long)blockIdx.y * TM;
    const long tn  = (long)blockIdx.x * TN;

    const _Float16* Ab = A  + (long)z*bsA;
    const _Float16* Bb = Bw + (long)z*bsB;

    v8f acc[4][2];
    #pragma unroll
    for (int i=0;i<4;i++)
        #pragma unroll
        for (int j=0;j<2;j++)
            #pragma unroll
            for (int r=0;r<8;r++) acc[i][j][r] = 0.0f;

    // staging: 256 threads, each owns one half-row (16 contiguous f16 = 2x16B)
    const int lrow = tid >> 1;          // 0..127
    const int lk0  = (tid & 1) << 4;    // 0 or 16
    const _Float16* pa = Ab + (tm + lrow)*lda + lk0;
    const _Float16* pb = Bb + (tn + lrow)*ldb + lk0;
    const int dso = lrow*LSTR + lk0;    // LDS element offset inside a buffer

    // fragment lane mapping per CDNA5 ISA (16-bit A 16x32 / B 32x16 layouts)
    const int mrow   = lane & 15;
    const int m_half = (lane >> 4) << 3;          // C/D rows 0-7 vs 8-15
    const int ka     = (lane < 16) ? 0 : 8;       // A K chunks {0-7,16-23}/{8-15,24-31}
    const int kb     = (lane < 16) ? 0 : 8;       // B K chunks {0-7,16-23}/{8-15,24-31}

    const int nk = K / TK;

    // prologue: tile 0 -> buffer 0 (4 async 16B transfers per thread)
    async_copy_b128(&As[0][dso],     pa);
    async_copy_b128(&As[0][dso + 8], pa + 8);
    async_copy_b128(&Bs[0][dso],     pb);
    async_copy_b128(&Bs[0][dso + 8], pb + 8);
    pa += TK; pb += TK;

    for (int t = 0; t < nk; ++t) {
        const int cur = t & 1;
        if (t + 1 < nk) {
            const int nxt = cur ^ 1;
            async_copy_b128(&As[nxt][dso],     pa);
            async_copy_b128(&As[nxt][dso + 8], pa + 8);
            async_copy_b128(&Bs[nxt][dso],     pb);
            async_copy_b128(&Bs[nxt][dso + 8], pb + 8);
            if (t + 2 < nk) {               // warm L2 for tile t+2
                __builtin_prefetch(pa + TK, 0, 1);
                __builtin_prefetch(pb + TK, 0, 1);
            }
            pa += TK; pb += TK;
            WAIT_ASYNC(4);                  // only newest 4 outstanding
        } else {
            WAIT_ASYNC(0);
        }
        __syncthreads();                    // tile t visible to all waves

        v16h af[4];
        #pragma unroll
        for (int i=0;i<4;i++){
            const _Float16* p = &As[cur][(wm*64 + i*16 + mrow)*LSTR + ka];
            union { v8h h[2]; v16h v; } u;
            u.h[0] = *(const v8h*)p;
            u.h[1] = *(const v8h*)(p + 16);
            af[i] = u.v;
        }
        v16h bf[2];
        #pragma unroll
        for (int j=0;j<2;j++){
            const _Float16* p = &Bs[cur][(wn*32 + j*16 + mrow)*LSTR + kb];
            union { v8h h[2]; v16h v; } u;
            u.h[0] = *(const v8h*)p;
            u.h[1] = *(const v8h*)(p + 16);
            bf[j] = u.v;
        }
        #pragma unroll
        for (int i=0;i<4;i++)
            #pragma unroll
            for (int j=0;j<2;j++)
                acc[i][j] = __builtin_amdgcn_wmma_f32_16x16x32_f16(
                    false, af[i], false, bf[j], (short)0, acc[i][j], false, false);
        __syncthreads();                    // buffer reusable by tile t+2 fill
    }

    const float sc = (EPI==3) ? scale[0] : 1.0f;
    #pragma unroll
    for (int i=0;i<4;i++){
        #pragma unroll
        for (int j=0;j<2;j++){
            const long gm0 = tm + wm*64 + i*16 + m_half;
            const long gn  = tn + wn*32 + j*16 + mrow;
            #pragma unroll
            for (int r=0;r<8;r++){
                const long gm = gm0 + r;
                float val = acc[i][j][r];
                if (EPI==1){ val += bias[gn]; val = (gn < 4096) ? silu_f(val) : gelu_f(val); }
                if (EPI==2){ val += bias[gn] + rowv[gm]*colv[gn]; val = sigmoid_f(val)*0.9f + 0.1f; }
                if (EPI==3){ val *= sc; }
                if (EPI==4 || EPI==6){ val += bias[gn] + resid[gm*ldr + gn]; }
                if (EPI==5){ val += bias[gn]; val = gelu_f(val); }
                const long ci = (long)z*bsC + gm*ldc + gn;
                if (EPI==0 || EPI==1 || EPI==2 || EPI==5)
                    ((_Float16*)Cout)[ci] = (_Float16)val;
                else
                    ((float*)Cout)[ci] = val;
            }
        }
    }
}

// --------------------------- elementwise kernels ---------------------------

// LayerNorm over last dim n; optional fused pos_info = dot(h, pe_mean)
__global__ __launch_bounds__(256)
void ln_fwd(const float* __restrict__ x, int n,
            const float* __restrict__ w, const float* __restrict__ b,
            _Float16* __restrict__ out,
            const float* __restrict__ pe_mean,
            float* __restrict__ pos_out)
{
    __shared__ float red[256];
    const int tid = threadIdx.x;
    const long row = blockIdx.x;
    const float* xr = x + row*(long)n;

    float s = 0.f;
    for (int i=tid;i<n;i+=256) s += xr[i];
    red[tid]=s; __syncthreads();
    for (int o=128;o>0;o>>=1){ if (tid<o) red[tid]+=red[tid+o]; __syncthreads(); }
    const float mean = red[0]/(float)n;
    __syncthreads();

    float vs = 0.f;
    for (int i=tid;i<n;i+=256){ float d = xr[i]-mean; vs += d*d; }
    red[tid]=vs; __syncthreads();
    for (int o=128;o>0;o>>=1){ if (tid<o) red[tid]+=red[tid+o]; __syncthreads(); }
    const float inv = rsqrtf(red[0]/(float)n + 1e-5f);
    __syncthreads();

    float pdot = 0.f;
    for (int i=tid;i<n;i+=256){
        float hv = (xr[i]-mean)*inv*w[i] + b[i];
        out[row*(long)n + i] = (_Float16)hv;
        if (pe_mean) pdot += hv*pe_mean[i];
    }
    if (pos_out){
        red[tid]=pdot; __syncthreads();
        for (int o=128;o>0;o>>=1){ if (tid<o) red[tid]+=red[tid+o]; __syncthreads(); }
        if (tid==0) pos_out[row] = red[0];
    }
}

// q' = q/||q|| + 0.1k ; k' = k/||k|| + 0.1q'   (row-wise L2 norms over E)
__global__ __launch_bounds__(256)
void qk_couple(const _Float16* __restrict__ qkv, long ldq,
               _Float16* __restrict__ qpre, _Float16* __restrict__ kpre, int E_)
{
    __shared__ float red[512];
    const int tid = threadIdx.x;
    const long row = blockIdx.x;
    const _Float16* qr = qkv + row*ldq;
    const _Float16* kr = qr + E_;

    float sq=0.f, sk=0.f;
    for (int i=tid;i<E_;i+=256){
        float q=(float)qr[i], k=(float)kr[i];
        sq += q*q; sk += k*k;
    }
    red[tid]=sq; red[256+tid]=sk; __syncthreads();
    for (int o=128;o>0;o>>=1){
        if (tid<o){ red[tid]+=red[tid+o]; red[256+tid]+=red[256+tid+o]; }
        __syncthreads();
    }
    const float nq = fmaxf(sqrtf(red[0]),   1e-12f);
    const float nk = fmaxf(sqrtf(red[256]), 1e-12f);
    for (int i=tid;i<E_;i+=256){
        float q=(float)qr[i], k=(float)kr[i];
        float qp = q/nq + 0.1f*k;
        float kp = k/nk + 0.1f*qp;
        qpre[row*(long)E_+i] = (_Float16)qp;
        kpre[row*(long)E_+i] = (_Float16)kp;
    }
}

// depthwise conv3 along L (+ optional fused beta multiply), optional
// transposed [b,e,l] output so downstream GEMMs stay K-contiguous
__global__ void dwconv(const _Float16* __restrict__ z, long ldz,
                       const float* __restrict__ w3,
                       const _Float16* __restrict__ beta,
                       _Float16* __restrict__ y,
                       int L_, int E_, int transposed, long total)
{
    long idx = (long)blockIdx.x*256 + threadIdx.x;
    if (idx >= total) return;
    const int  e  = (int)(idx % E_);
    const long bl = idx / E_;
    const int  l  = (int)(bl % L_);
    const long b  = bl / L_;

    const _Float16* zr = z + bl*ldz + e;
    float acc = w3[e*3+1]*(float)zr[0];
    if (l > 0)    acc += w3[e*3+0]*(float)zr[-ldz];
    if (l < L_-1) acc += w3[e*3+2]*(float)zr[ldz];
    if (beta) acc *= (float)beta[bl*(long)E_ + e];
    const long oidx = transposed ? ((b*E_ + e)*(long)L_ + l) : (bl*(long)E_ + e);
    y[oidx] = (_Float16)acc;
}

__global__ void f32_to_f16(const float* __restrict__ in, _Float16* __restrict__ out, long n)
{
    long i = (long)blockIdx.x*256 + threadIdx.x;
    if (i < n) out[i] = (_Float16)in[i];
}

// split w_beta into f16 first-H block + f32 rowsum of second-H block
__global__ __launch_bounds__(256)
void wbeta_prep(const float* __restrict__ wb, _Float16* __restrict__ w16,
                float* __restrict__ rowsum, int H_)
{
    __shared__ float red[256];
    const int tid = threadIdx.x;
    const long e = blockIdx.x;
    const float* r = wb + e*(long)(2*H_);
    float s = 0.f;
    for (int i=tid;i<H_;i+=256){
        w16[e*(long)H_ + i] = (_Float16)r[i];
        s += r[H_ + i];
    }
    red[tid]=s; __syncthreads();
    for (int o=128;o>0;o>>=1){ if (tid<o) red[tid]+=red[tid+o]; __syncthreads(); }
    if (tid==0) rowsum[e] = red[0];
}

// pe_mean[h] = mean over first L rows of rel_pos
__global__ void pe_mean_k(const float* __restrict__ rel, float* __restrict__ pm,
                          int H_, int Lr)
{
    int h = blockIdx.x*256 + threadIdx.x;
    if (h >= H_) return;
    float s = 0.f;
    for (int m=0;m<Lr;m++) s += rel[(long)m*H_ + h];
    pm[h] = s/(float)Lr;
}

// ------------------------------- launcher ----------------------------------
extern "C" void kernel_launch(void* const* d_in, const int* in_sizes, int n_in,
                              void* d_out, int out_size, void* d_ws, size_t ws_size,
                              hipStream_t stream)
{
    const float* x       = (const float*)d_in[0];
    const float* ln1_w   = (const float*)d_in[1];
    const float* ln1_b   = (const float*)d_in[2];
    const float* ln2_w   = (const float*)d_in[3];
    const float* ln2_b   = (const float*)d_in[4];
    const float* w_qkv   = (const float*)d_in[5];
    const float* b_qkv   = (const float*)d_in[6];
    const float* w_out   = (const float*)d_in[7];
    const float* b_out   = (const float*)d_in[8];
    const float* rel_pos = (const float*)d_in[9];
    const float* w_beta  = (const float*)d_in[10];
    const float* b_beta  = (const float*)d_in[11];
    const float* w1      = (const float*)d_in[12];
    const float* b1      = (const float*)d_in[13];
    const float* w2      = (const float*)d_in[14];
    const float* b2      = (const float*)d_in[15];
    const float* conv_w  = (const float*)d_in[16];
    const float* ascale  = (const float*)d_in[17];

    char* ws = (char*)d_ws;
    size_t off = 0;
    auto take = [&](size_t bytes)->char*{
        char* p = ws + off; off += (bytes + 255) & ~(size_t)255; return p;
    };

    float*    pe_m    = (float*)   take((size_t)Hn*4);
    float*    posi    = (float*)   take((size_t)BLn*4);
    _Float16* h16     = (_Float16*)take((size_t)BLn*Hn*2);
    _Float16* wqkv16  = (_Float16*)take((size_t)3*En*Hn*2);
    _Float16* wbeta16 = (_Float16*)take((size_t)En*Hn*2);
    float*    wbrs    = (float*)   take((size_t)En*4);
    _Float16* wout16  = (_Float16*)take((size_t)Hn*En*2);
    _Float16* w1h     = (_Float16*)take((size_t)4*Hn*Hn*2);
    _Float16* w2h     = (_Float16*)take((size_t)Hn*4*Hn*2);
    _Float16* qkvact  = (_Float16*)take((size_t)BLn*3*En*2);
    _Float16* qpre    = (_Float16*)take((size_t)BLn*En*2);
    _Float16* kpre    = (_Float16*)take((size_t)BLn*En*2);
    _Float16* q16     = (_Float16*)take((size_t)BLn*En*2);
    _Float16* kT      = (_Float16*)take((size_t)BLn*En*2);   // [b,e,l]
    _Float16* vnT     = (_Float16*)take((size_t)BLn*En*2);   // [b,e,l]
    _Float16* beta16  = (_Float16*)take((size_t)BLn*En*2);
    _Float16* state16 = (_Float16*)take((size_t)Bn*En*En*2); // G[f,e]
    float*    oattn   = (float*)   take((size_t)BLn*En*4);
    _Float16* oln16   = (_Float16*)take((size_t)BLn*En*2);
    float*    xnew    = (float*)   take((size_t)BLn*Hn*4);
    _Float16* h2h     = (_Float16*)take((size_t)BLn*Hn*2);
    _Float16* mid16   = (_Float16*)take((size_t)BLn*4*Hn*2);

    // one-time (per launch) weight conversion f32 -> f16
    { long n=(long)3*En*Hn; f32_to_f16<<<dim3((unsigned)((n+255)/256)),dim3(256),0,stream>>>(w_qkv, wqkv16, n); }
    { long n=(long)Hn*En;   f32_to_f16<<<dim3((unsigned)((n+255)/256)),dim3(256),0,stream>>>(w_out, wout16, n); }
    { long n=(long)4*Hn*Hn; f32_to_f16<<<dim3((unsigned)((n+255)/256)),dim3(256),0,stream>>>(w1, w1h, n); }
    { long n=(long)Hn*4*Hn; f32_to_f16<<<dim3((unsigned)((n+255)/256)),dim3(256),0,stream>>>(w2, w2h, n); }
    wbeta_prep<<<dim3(En),dim3(256),0,stream>>>(w_beta, wbeta16, wbrs, Hn);
    pe_mean_k <<<dim3(Hn/256),dim3(256),0,stream>>>(rel_pos, pe_m, Hn, Ln);

    // LN1 (+ fused pos_info dot)
    ln_fwd<<<dim3(BLn),dim3(256),0,stream>>>(x, Hn, ln1_w, ln1_b, h16, pe_m, posi);

    // qkv = h @ w_qkv^T + b ; silu/silu/gelu fused
    gemm_wmma<1><<<dim3(3*En/TN, BLn/TM, 1),dim3(256),0,stream>>>(
        h16, Hn, 0, wqkv16, Hn, 0, Hn,
        b_qkv, nullptr, nullptr, nullptr, nullptr, 0,
        qkvact, 3*En, 0);

    // beta = sigmoid(h @ Wb1^T + pos*rowsum + b)*0.9 + 0.1
    gemm_wmma<2><<<dim3(En/TN, BLn/TM, 1),dim3(256),0,stream>>>(
        h16, Hn, 0, wbeta16, Hn, 0, Hn,
        b_beta, posi, wbrs, nullptr, nullptr, 0,
        beta16, En, 0);

    // coupled q/k normalization
    qk_couple<<<dim3(BLn),dim3(256),0,stream>>>(qkvact, 3*En, qpre, kpre, En);

    // depthwise conv (q normal layout; k,v transposed; v fused with beta)
    const long tot = (long)BLn*En;
    dwconv<<<dim3((unsigned)(tot/256)),dim3(256),0,stream>>>(qpre, En,   conv_w, nullptr, q16, Ln, En, 0, tot);
    dwconv<<<dim3((unsigned)(tot/256)),dim3(256),0,stream>>>(kpre, En,   conv_w, nullptr, kT,  Ln, En, 1, tot);
    dwconv<<<dim3((unsigned)(tot/256)),dim3(256),0,stream>>>(qkvact+2*En, 3*En, conv_w, beta16, vnT, Ln, En, 1, tot);

    // G[f,e] = sum_l v_new[l,f] * k[l,e]   (per batch)
    gemm_wmma<0><<<dim3(En/TN, En/TM, Bn),dim3(256),0,stream>>>(
        vnT, Ln, (long)En*Ln, kT, Ln, (long)En*Ln, Ln,
        nullptr, nullptr, nullptr, nullptr, nullptr, 0,
        state16, En, (long)En*En);

    // out[l,f] = (q[l,:] . G[f,:]) * attn_scale   (per batch)
    gemm_wmma<3><<<dim3(En/TN, Ln/TM, Bn),dim3(256),0,stream>>>(
        q16, En, (long)Ln*En, state16, En, (long)En*En, En,
        nullptr, nullptr, nullptr, ascale, nullptr, 0,
        oattn, En, (long)Ln*En);

    // LN2 -> f16
    ln_fwd<<<dim3(BLn),dim3(256),0,stream>>>(oattn, En, ln2_w, ln2_b, oln16, nullptr, nullptr);

    // x_new = x + LN2(out) @ w_out^T + b_out
    gemm_wmma<4><<<dim3(Hn/TN, BLn/TM, 1),dim3(256),0,stream>>>(
        oln16, En, 0, wout16, En, 0, En,
        b_out, nullptr, nullptr, nullptr, x, Hn,
        xnew, Hn, 0);

    // MLP
    ln_fwd<<<dim3(BLn),dim3(256),0,stream>>>(xnew, Hn, ln1_w, ln1_b, h2h, nullptr, nullptr);

    gemm_wmma<5><<<dim3(4*Hn/TN, BLn/TM, 1),dim3(256),0,stream>>>(
        h2h, Hn, 0, w1h, Hn, 0, Hn,
        b1, nullptr, nullptr, nullptr, nullptr, 0,
        mid16, 4*Hn, 0);

    gemm_wmma<6><<<dim3(Hn/TN, BLn/TM, 1),dim3(256),0,stream>>>(
        mid16, 4*Hn, 0, w2h, 4*Hn, 0, 4*Hn,
        b2, nullptr, nullptr, nullptr, xnew, Hn,
        (float*)d_out, Hn, 0);

    (void)in_sizes; (void)n_in; (void)out_size; (void)ws_size;
}
